// HybridRefineBlock_51805895524536
// MI455X (gfx1250) — matrix operand order repaired and checked by reading, not compiled, not run on hardware
//
#include <hip/hip_runtime.h>

// ---------------------------------------------------------------------------
// Mamba stack (NL=4, B=1, L=1024, D=1024, DI=2048, N=16, R=64, K=4)
// GEMMs: v_wmma_f32_16x16x32_bf16, 64x64-per-wave register tiles, bf16 weights.
// Scan: one thread per (channel, state) pair, shuffle reduction over n,
//       B/C staged to LDS with global_load_async_to_lds_b128 (ASYNCcnt).
// ---------------------------------------------------------------------------

typedef __bf16 bf16;
typedef __attribute__((ext_vector_type(16))) __bf16 v16bf;
typedef __attribute__((ext_vector_type(8)))  __bf16 v8bf;
typedef __attribute__((ext_vector_type(8)))  float  v8f;

constexpr int NL = 4;
constexpr int D  = 1024;
constexpr int DI = 2048;
constexpr int SN = 16;    // state size N
constexpr int R  = 64;
constexpr int CK = 4;     // conv kernel width
constexpr int L  = 1024;
constexpr int E2 = 2 * DI;        // 4096
constexpr int XP = R + 2 * SN;    // 96

// ---------------------------------------------------------------------------
// WMMA fragment helpers (contiguous k order used here, wave32):
//  A 16x32: lane l (row=l&15, hi=l>>4) holds K {hi*16..+15} (one 32B run)
//  B 32x16: lane l (col=l&15, hi=l>>4) holds K {hi*16..+15} (one 32B run)
//  C/D f32 16x16: VGPR v -> row v+8*hi, col lane&15.
// ---------------------------------------------------------------------------

__device__ __forceinline__ v16bf load_a_frag(const bf16* p) {
  v8bf lo = *(const v8bf*)(p);
  v8bf hi = *(const v8bf*)(p + 8);
  return __builtin_shufflevector(lo, hi, 0, 1, 2, 3, 4, 5, 6, 7,
                                 8, 9, 10, 11, 12, 13, 14, 15);
}

template <int MT, int NT>
__device__ __forceinline__ void wmma_block(const bf16* __restrict__ A, int lda,
                                           const bf16* __restrict__ W, int ldw,
                                           int m0, int n0, int Kdim,
                                           v8f acc[MT][NT]) {
  const int lane = threadIdx.x & 31;
  const int row  = lane & 15;
  const int hi   = lane >> 4;
  const bf16* abase = A + (size_t)(m0 + row) * lda + hi * 16;
  const bf16* wbase = W + (size_t)(n0 + row) * ldw + hi * 16;
  for (int k = 0; k < Kdim; k += 32) {
    v16bf afrag[MT];
#pragma unroll
    for (int mi = 0; mi < MT; ++mi)
      afrag[mi] = load_a_frag(abase + (size_t)mi * 16 * lda + k);
#pragma unroll
    for (int j = 0; j < NT; ++j) {
      v16bf b = *(const v16bf*)(wbase + (size_t)j * 16 * ldw + k);
#pragma unroll
      for (int mi = 0; mi < MT; ++mi)
        acc[mi][j] = __builtin_amdgcn_wmma_f32_16x16x32_bf16(
            false, afrag[mi], false, b, (short)0, acc[mi][j], false, false);
    }
  }
}

__device__ __forceinline__ void store_tile(float* __restrict__ C, int ldc,
                                           int m0, int n0, v8f acc) {
  const int lane = threadIdx.x & 31;
  const int col  = lane & 15;
  const int hi   = lane >> 4;
  float* cp = C + (size_t)(m0 + hi * 8) * ldc + n0 + col;
#pragma unroll
  for (int v = 0; v < 8; ++v) cp[(size_t)v * ldc] = acc[v];
}

// ---------------------------------------------------------------------------
// GEMM kernels (wave = 64x64 output tile except gemm_xp)
// ---------------------------------------------------------------------------

// xz[L,4096] = xn_bf[L,1024] @ in_w_bf[4096,1024]^T
__global__ __launch_bounds__(256) void gemm_xz_k(const bf16* __restrict__ A,
                                                 const bf16* __restrict__ W,
                                                 float* __restrict__ C) {
  const int gw = blockIdx.x * 8 + (threadIdx.x >> 5);
  constexpr int TN = E2 / 64;  // 64
  const int m0 = (gw / TN) * 64;
  const int n0 = (gw % TN) * 64;
  v8f acc[4][4] = {};
  wmma_block<4, 4>(A, D, W, D, m0, n0, D, acc);
#pragma unroll
  for (int mi = 0; mi < 4; ++mi)
#pragma unroll
    for (int j = 0; j < 4; ++j)
      store_tile(C, E2, m0 + 16 * mi, n0 + 16 * j, acc[mi][j]);
}

// xp[L,96] = xc_bf[L,2048] @ xproj_w_bf[96,2048]^T   (4x1 tiles, B reuse x4)
__global__ __launch_bounds__(256) void gemm_xp_k(const bf16* __restrict__ A,
                                                 const bf16* __restrict__ W,
                                                 float* __restrict__ C) {
  const int gw = blockIdx.x * 8 + (threadIdx.x >> 5);
  const int m0 = (gw / 6) * 64;
  const int n0 = (gw % 6) * 16;
  v8f acc[4][1] = {};
  wmma_block<4, 1>(A, DI, W, DI, m0, n0, DI, acc);
#pragma unroll
  for (int mi = 0; mi < 4; ++mi)
    store_tile(C, XP, m0 + 16 * mi, n0, acc[mi][0]);
}

// delta[L,2048] = softplus(xp_bf[L,64] @ dt_w_bf[2048,64]^T + dt_b)
__global__ __launch_bounds__(256) void gemm_delta_k(const bf16* __restrict__ A,
                                                    const bf16* __restrict__ W,
                                                    const float* __restrict__ bias,
                                                    float* __restrict__ C) {
  const int gw = blockIdx.x * 8 + (threadIdx.x >> 5);
  constexpr int TN = DI / 64;  // 32
  const int m0 = (gw / TN) * 64;
  const int n0 = (gw % TN) * 64;
  v8f acc[4][4] = {};
  wmma_block<4, 4>(A, R, W, R, m0, n0, R, acc);
  const int lane = threadIdx.x & 31;
  const int col  = lane & 15;
  const int hi   = lane >> 4;
#pragma unroll
  for (int mi = 0; mi < 4; ++mi)
#pragma unroll
    for (int j = 0; j < 4; ++j) {
      const int c = n0 + 16 * j + col;
      const float b = bias[c];
      float* cp = C + (size_t)(m0 + 16 * mi + hi * 8) * DI + c;
#pragma unroll
      for (int v = 0; v < 8; ++v) {
        float t = acc[mi][j][v] + b;
        cp[(size_t)v * DI] = (t > 20.f) ? t : log1pf(__expf(t));
      }
    }
}

// x_cur[L,1024] += y_bf[L,2048] @ out_w_bf[1024,2048]^T   (residual fused)
__global__ __launch_bounds__(256) void gemm_out_k(const bf16* __restrict__ A,
                                                  const bf16* __restrict__ W,
                                                  float* __restrict__ C) {
  const int gw = blockIdx.x * 8 + (threadIdx.x >> 5);
  constexpr int TN = D / 64;  // 16
  const int m0 = (gw / TN) * 64;
  const int n0 = (gw % TN) * 64;
  v8f acc[4][4] = {};
  wmma_block<4, 4>(A, DI, W, DI, m0, n0, DI, acc);
  const int lane = threadIdx.x & 31;
  const int col  = lane & 15;
  const int hi   = lane >> 4;
#pragma unroll
  for (int mi = 0; mi < 4; ++mi)
#pragma unroll
    for (int j = 0; j < 4; ++j) {
      float* cp = C + (size_t)(m0 + 16 * mi + hi * 8) * D + n0 + 16 * j + col;
#pragma unroll
      for (int v = 0; v < 8; ++v) cp[(size_t)v * D] += acc[mi][j][v];
    }
}

// ---------------------------------------------------------------------------
// fp32 -> bf16 weight conversion (per layer, once per launch)
// ---------------------------------------------------------------------------

__global__ __launch_bounds__(256) void cvt_w_k(const float* __restrict__ src,
                                               bf16* __restrict__ dst, int n) {
  const int i = blockIdx.x * 256 + threadIdx.x;
  if (i < n) dst[i] = (bf16)src[i];
}

// ---------------------------------------------------------------------------
// RMSNorm
// ---------------------------------------------------------------------------

__global__ __launch_bounds__(256) void rmsnorm_bf16_k(const float* __restrict__ x,
                                                      const float* __restrict__ w,
                                                      bf16* __restrict__ out) {
  const int l = blockIdx.x, tid = threadIdx.x;
  __shared__ float red[256];
  const float* xr = x + (size_t)l * D;
  float v[4], s = 0.f;
#pragma unroll
  for (int j = 0; j < 4; ++j) { v[j] = xr[tid + 256 * j]; s += v[j] * v[j]; }
  red[tid] = s;
  __syncthreads();
  for (int off = 128; off > 0; off >>= 1) {
    if (tid < off) red[tid] += red[tid + off];
    __syncthreads();
  }
  const float rs = rsqrtf(red[0] * (1.f / D) + 1e-6f);
#pragma unroll
  for (int j = 0; j < 4; ++j)
    out[(size_t)l * D + tid + 256 * j] = (bf16)(v[j] * rs * w[tid + 256 * j]);
}

__global__ __launch_bounds__(256) void rmsnorm_final_k(const float* __restrict__ x,
                                                       const float* __restrict__ w,
                                                       float* __restrict__ out) {
  const int l = blockIdx.x, tid = threadIdx.x;
  __shared__ float red[256];
  const float* xr = x + (size_t)l * D;
  float v[4], s = 0.f;
#pragma unroll
  for (int j = 0; j < 4; ++j) { v[j] = xr[tid + 256 * j]; s += v[j] * v[j]; }
  red[tid] = s;
  __syncthreads();
  for (int off = 128; off > 0; off >>= 1) {
    if (tid < off) red[tid] += red[tid + off];
    __syncthreads();
  }
  const float rs = rsqrtf(red[0] * (1.f / D) + 1e-6f);
#pragma unroll
  for (int j = 0; j < 4; ++j)
    out[(size_t)l * D + tid + 256 * j] = v[j] * rs * w[tid + 256 * j];
}

// ---------------------------------------------------------------------------
// Depthwise causal conv (K=4) + bias + silu; writes fp32 and bf16 copies
// ---------------------------------------------------------------------------

__global__ __launch_bounds__(256) void conv_silu_k(const float* __restrict__ xz,
                                                   const float* __restrict__ cw,
                                                   const float* __restrict__ cb,
                                                   float* __restrict__ xc,
                                                   bf16* __restrict__ xc_bf) {
  const int i = blockIdx.x * 256 + threadIdx.x;
  const int l = i / DI;
  const int d = i - l * DI;
  float acc = cb[d];
#pragma unroll
  for (int k = 0; k < CK; ++k) {
    const int lk = l - (CK - 1) + k;
    if (lk >= 0) acc += cw[d * CK + k] * xz[(size_t)lk * E2 + d];
  }
  const float s = acc / (1.f + __expf(-acc));
  xc[i] = s;
  xc_bf[i] = (bf16)s;
}

__global__ __launch_bounds__(256) void cvt_xp_k(const float* __restrict__ xp,
                                                bf16* __restrict__ xp_bf) {
  const int i = blockIdx.x * 256 + threadIdx.x;  // i in [0, L*R)
  const int l = i >> 6;
  const int r = i & 63;
  xp_bf[i] = (bf16)xp[(size_t)l * XP + r];
}

// ---------------------------------------------------------------------------
// Selective scan: one thread per (channel d, state n). 16x more parallelism
// than thread-per-channel: 1 exp per thread per step; y = sum over n via 4
// __shfl_xor hops within a 16-lane group (wave32).
// B/C stream staged to LDS via CDNA5 async-to-LDS (ASYNCcnt) loads.
// ---------------------------------------------------------------------------

constexpr int SCHUNK = 64;

__global__ __launch_bounds__(256) void scan_k(const float* __restrict__ delta,
                                              const float* __restrict__ xc,
                                              const float* __restrict__ xp,
                                              const float* __restrict__ xz,
                                              const float* __restrict__ A_log,
                                              const float* __restrict__ Dp,
                                              bf16* __restrict__ y_bf) {
  const int tid = threadIdx.x;
  const int n   = tid & 15;                   // state index within channel
  const int d   = blockIdx.x * 16 + (tid >> 4);
  const float A = -__expf(A_log[(size_t)d * SN + n]);
  const float Dd = Dp[d];
  float h = 0.f;

  // sBC[tt][0..15] = B(t,n), sBC[tt][16..31] = C(t,n)
  __shared__ float sBC[SCHUNK][32];
  const unsigned lds_base = (unsigned)(uintptr_t)&sBC[0][0];

  for (int t0 = 0; t0 < L; t0 += SCHUNK) {
    // Async-copy the 8KB B/C chunk: 512 x 16B transfers, ASYNCcnt tracked.
    for (int i = tid; i < SCHUNK * 8; i += 256) {
      const int tt = i >> 3;
      const int q  = i & 7;
      const float* g = xp + (size_t)(t0 + tt) * XP + R + q * 4;
      const unsigned lds = lds_base + (unsigned)i * 16u;
      asm volatile("global_load_async_to_lds_b128 %0, %1, off"
                   :: "v"(lds), "v"(g) : "memory");
    }
    asm volatile("s_wait_asynccnt 0x0" ::: "memory");
    __syncthreads();

    for (int tt = 0; tt < SCHUNK; ++tt) {
      const int t = t0 + tt;
      const float dl  = delta[(size_t)t * DI + d];   // broadcast across 16 lanes
      const float xcv = xc[(size_t)t * DI + d];
      const float a = __expf(dl * A);
      h = a * h + (dl * xcv) * sBC[tt][n];
      float p = h * sBC[tt][16 + n];
      // reduce over the 16-lane n-group
      p += __shfl_xor(p, 8);
      p += __shfl_xor(p, 4);
      p += __shfl_xor(p, 2);
      p += __shfl_xor(p, 1);
      if (n == 0) {
        const float y = p + xcv * Dd;
        const float zv = xz[(size_t)t * E2 + DI + d];
        const float g = zv / (1.f + __expf(-zv));
        y_bf[(size_t)t * DI + d] = (bf16)(y * g);
      }
    }
    __syncthreads();
  }
}

// ---------------------------------------------------------------------------
// Host orchestration
// ---------------------------------------------------------------------------

extern "C" void kernel_launch(void* const* d_in, const int* in_sizes, int n_in,
                              void* d_out, int out_size, void* d_ws, size_t ws_size,
                              hipStream_t stream) {
  const float* x       = (const float*)d_in[0];
  const float* norm_w  = (const float*)d_in[1];
  const float* in_w    = (const float*)d_in[2];
  const float* conv_w  = (const float*)d_in[3];
  const float* conv_b  = (const float*)d_in[4];
  const float* A_log   = (const float*)d_in[5];
  const float* Dp      = (const float*)d_in[6];
  const float* xproj_w = (const float*)d_in[7];
  const float* dt_w    = (const float*)d_in[8];
  const float* dt_b    = (const float*)d_in[9];
  const float* out_w   = (const float*)d_in[10];
  const float* final_w = (const float*)d_in[11];

  char* ws = (char*)d_ws;
  size_t off = 0;
  auto take = [&](size_t bytes) {
    char* p = ws + off;
    off += (bytes + 255) & ~(size_t)255;
    return p;
  };
  float* x_cur  = (float*)take((size_t)L * D * 4);
  bf16*  xn_bf  = (bf16*) take((size_t)L * D * 2);
  float* xz     = (float*)take((size_t)L * E2 * 4);
  float* xc     = (float*)take((size_t)L * DI * 4);
  bf16*  xc_bf  = (bf16*) take((size_t)L * DI * 2);
  float* xp     = (float*)take((size_t)L * XP * 4);
  bf16*  xp_bf  = (bf16*) take((size_t)L * R * 2);
  float* delta  = (float*)take((size_t)L * DI * 4);
  bf16*  y_bf   = (bf16*) take((size_t)L * DI * 2);
  bf16*  w_in   = (bf16*) take((size_t)E2 * D * 2);   // per-layer bf16 weights
  bf16*  w_xp   = (bf16*) take((size_t)XP * DI * 2);
  bf16*  w_dt   = (bf16*) take((size_t)DI * R * 2);
  bf16*  w_out  = (bf16*) take((size_t)D * DI * 2);

  hipMemcpyAsync(x_cur, x, (size_t)L * D * sizeof(float),
                 hipMemcpyDeviceToDevice, stream);

  for (int i = 0; i < NL; ++i) {
    // per-layer weight conversion fp32 -> bf16
    cvt_w_k<<<(E2 * D) / 256, 256, 0, stream>>>(
        in_w + (size_t)i * E2 * D, w_in, E2 * D);
    cvt_w_k<<<(XP * DI) / 256, 256, 0, stream>>>(
        xproj_w + (size_t)i * XP * DI, w_xp, XP * DI);
    cvt_w_k<<<(DI * R) / 256, 256, 0, stream>>>(
        dt_w + (size_t)i * DI * R, w_dt, DI * R);
    cvt_w_k<<<(D * DI) / 256, 256, 0, stream>>>(
        out_w + (size_t)i * D * DI, w_out, D * DI);

    rmsnorm_bf16_k<<<L, 256, 0, stream>>>(x_cur, norm_w + (size_t)i * D, xn_bf);
    gemm_xz_k<<<((L / 64) * (E2 / 64)) / 8, 256, 0, stream>>>(xn_bf, w_in, xz);
    conv_silu_k<<<(L * DI) / 256, 256, 0, stream>>>(
        xz, conv_w + (size_t)i * DI * CK, conv_b + (size_t)i * DI, xc, xc_bf);
    gemm_xp_k<<<((L / 64) * 6) / 8, 256, 0, stream>>>(xc_bf, w_xp, xp);
    cvt_xp_k<<<(L * R) / 256, 256, 0, stream>>>(xp, xp_bf);
    gemm_delta_k<<<((L / 64) * (DI / 64)) / 8, 256, 0, stream>>>(
        xp_bf, w_dt, dt_b + (size_t)i * DI, delta);
    scan_k<<<DI / 16, 256, 0, stream>>>(
        delta, xc, xp, xz, A_log + (size_t)i * DI * SN, Dp + (size_t)i * DI, y_bf);
    gemm_out_k<<<((L / 64) * (D / 64)) / 8, 256, 0, stream>>>(y_bf, w_out, x_cur);
  }
  rmsnorm_final_k<<<L, 256, 0, stream>>>(x_cur, final_w, (float*)d_out);
}
